// Dynamic_LSQConv2d_901943132743
// MI455X (gfx1250) — compile-verified, matrix operand order checked by reading, not run on hardware
//
#include <hip/hip_runtime.h>
#include <hip/hip_bf16.h>
#include <stdint.h>

// ---------------------------------------------------------------------------
// Dynamic LSQ-quantized mixture-of-experts conv, MI455X (gfx1250) wave32.
// Heavy path = per-sample 3x3 conv as implicit GEMM on
// v_wmma_f32_16x16x32_bf16 with bf16-hi/lo split emulation of fp32,
// weight tiles staged LDS-side by the Tensor Data Mover (TDM).
// ---------------------------------------------------------------------------

typedef __attribute__((ext_vector_type(16))) __bf16 v16bf;
typedef __attribute__((ext_vector_type(8)))  float  v8f;

#define B_    32
#define C_    256
#define OC_   256
#define HID_  65
#define KEXP  4
#define HW_   3136
#define W_IMG 56
#define MSIZE (OC_ * C_ * 9)              // 589824 elements per expert
#define RAW_OFF ((size_t)B_ * OC_ * HW_)  // 25690112 floats, then raw (B,4)

__device__ __forceinline__ uint32_t f32_to_bf16(float f) {
    uint32_t x = __float_as_uint(f);
    return ((x + 0x7FFFu + ((x >> 16) & 1u)) >> 16) & 0xFFFFu;  // RNE
}
__device__ __forceinline__ float bf16hi_to_f32(uint32_t h) {
    return __uint_as_float(h << 16);
}

// ---------------------------------------------------------------------------
// TDM descriptor build + issue (ISA cdna5 §8): 2D bf16 tile, 128 rows x 32
// elements, row stride 256 elements. This toolchain exposes the 6-argument
// builtin: (g0 u32x4, g1 i32x8, g2 i32x4, g3 i32x4, extra i32x8, cpol).
// ---------------------------------------------------------------------------
#if __has_builtin(__builtin_amdgcn_tensor_load_to_lds)
#define HAVE_TDM 1
typedef __attribute__((ext_vector_type(4))) unsigned int tdm_v4u;
typedef __attribute__((ext_vector_type(8))) int         tdm_v8i;
typedef __attribute__((ext_vector_type(4))) int         tdm_v4i;

__device__ __forceinline__ void tdm_load_tile_bf16(const uint16_t* gsrc,
                                                   uint32_t lds_off) {
    const uint64_t ga = (uint64_t)(uintptr_t)gsrc;
    tdm_v4u g0;
    g0[0] = 1u;                                    // count=1, user-mode load
    g0[1] = lds_off;                               // lds_addr (bytes)
    g0[2] = (uint32_t)ga;                          // global_addr[31:0]
    g0[3] = (uint32_t)((ga >> 32) & 0x01FFFFFFu)   // global_addr[56:32]
          | (2u << 30);                            // type = 2 ("image")
    tdm_v8i g1;
    g1[0] = (int)(1u << 16);      // wg_mask=0, data_size=1 (2 bytes)
    g1[1] = (int)(32u << 16);     // tensor_dim0[15:0]=32 (bits 63:48)
    g1[2] = (int)(128u << 16);    // tensor_dim1[15:0]=128 (bits 95:80)
    g1[3] = (int)(32u << 16);     // tile_dim0=32 (bits 127:112)
    g1[4] = 128;                  // tile_dim1=128 (bits 143:128)
    g1[5] = 256;                  // tensor_dim0_stride=256 elems (bits 191:160)
    g1[6] = 0;                    // stride hi / dim1_stride lo
    g1[7] = 0;
    tdm_v4i gz4 = {0, 0, 0, 0};
    tdm_v8i gz8 = {0, 0, 0, 0, 0, 0, 0, 0};
    __builtin_amdgcn_tensor_load_to_lds(g0, g1, gz4, gz4, gz8, 0);
}
#endif

// ---------------------------------------------------------------------------
// Kernel 1: attention head. pooled -> fc1+relu -> fc2 -> raw ; expert=argmax.
// ---------------------------------------------------------------------------
__global__ __launch_bounds__(256) void attn_kernel(
    const float* __restrict__ x, const float* __restrict__ w_fc1,
    const float* __restrict__ w_fc2, const float* __restrict__ b_fc2,
    float* __restrict__ raw_out, int* __restrict__ expert_idx)
{
    __shared__ float pool[C_];
    __shared__ float hbuf[HID_];
    __shared__ float rawbuf[KEXP];
    const int b = blockIdx.x;
    const int t = threadIdx.x, lane = t & 31, wave = t >> 5;
    const float* xb = x + (size_t)b * C_ * HW_;

    for (int ci = 0; ci < 32; ++ci) {
        const int c = wave * 32 + ci;
        const float* xc = xb + (size_t)c * HW_;
        float s = 0.f;
        for (int j = lane; j < HW_; j += 32) s += xc[j];
        for (int off = 16; off > 0; off >>= 1) s += __shfl_xor(s, off, 32);
        if (lane == 0) pool[c] = s * (1.0f / HW_);
    }
    __syncthreads();
    if (t < HID_) {
        float acc = 0.f;
        for (int c = 0; c < C_; ++c) acc += pool[c] * w_fc1[t * C_ + c];
        hbuf[t] = acc > 0.f ? acc : 0.f;
    }
    __syncthreads();
    if (t < KEXP) {
        float acc = b_fc2[t];
        for (int j = 0; j < HID_; ++j) acc += hbuf[j] * w_fc2[t * HID_ + j];
        rawbuf[t] = acc;
        raw_out[b * KEXP + t] = acc;
    }
    __syncthreads();
    if (t == 0) {  // first-max argmax, like jnp.argmax
        int best = 0; float bv = rawbuf[0];
        for (int k = 1; k < KEXP; ++k) if (rawbuf[k] > bv) { bv = rawbuf[k]; best = k; }
        expert_idx[b] = best;
    }
}

// ---------------------------------------------------------------------------
// Kernel 2: LSQ quantize + split fp32 -> bf16 hi/lo, relayout to
// [expert][rs][oc][c] so GEMM A-tiles are contiguous TDM 2D tiles.
// ---------------------------------------------------------------------------
__global__ __launch_bounds__(256) void quant_split_kernel(
    const float* __restrict__ weight, const float* __restrict__ alpha,
    uint16_t* __restrict__ wq_hi, uint16_t* __restrict__ wq_lo)
{
    const int i = blockIdx.x * 256 + threadIdx.x;
    if (i >= KEXP * MSIZE) return;
    const int k = i / MSIZE;
    const int m = i - k * MSIZE;
    const float a = alpha[k];
    float d = weight[i] / a;
    if (m == 0) {  // reference clamps only element [k,0] for k in {0,1,2}
        if (k == 0)      d = fminf(fmaxf(d, -2.f), 1.f);
        else if (k == 1) d = fminf(fmaxf(d, -4.f), 3.f);
        else if (k == 2) d = fminf(fmaxf(d, -8.f), 7.f);
    }
    const float wq = rintf(d) * a;  // RNE round like jnp.round
    const uint32_t hi = f32_to_bf16(wq);
    const float    lo = wq - bf16hi_to_f32(hi);
    const int oc = m / (C_ * 9);
    const int c  = (m / 9) % C_;
    const int rs = m % 9;
    const size_t dst = (((size_t)k * 9 + rs) * OC_ + oc) * C_ + c;
    wq_hi[dst] = (uint16_t)hi;
    wq_lo[dst] = (uint16_t)f32_to_bf16(lo);
}

// ---------------------------------------------------------------------------
// Kernel 3: implicit-GEMM conv. Block tile 128(oc) x 128(p); 8 waves as 2x4;
// wave tile 64x32 = 4x2 accumulators of 16x16 f32. K = 9 taps x 256 ch.
// bf16x3: acc += Ah*Bh + Ah*Bl + Al*Bh (f32 accumulate).
// A tiles via TDM (overlapped with im2col B staging), B tiles split on the fly.
// ---------------------------------------------------------------------------
__global__ __launch_bounds__(256) void conv_wmma_kernel(
    const float* __restrict__ x,
    const uint16_t* __restrict__ wq_hi, const uint16_t* __restrict__ wq_lo,
    const float* __restrict__ bias_w, const int* __restrict__ expert_idx,
    float* __restrict__ out)
{
    __shared__ uint16_t sAh[128 * 32], sAl[128 * 32];   // [oc][c]  hi/lo
    __shared__ uint16_t sBh[128 * 32], sBl[128 * 32];   // [p][c]   hi/lo

    const int nt = blockIdx.x;          // p-tile   (0..24, 25*128 >= 3136)
    const int mt = blockIdx.y;          // oc-tile  (0..1)
    const int b  = blockIdx.z;
    const int t = threadIdx.x, lane = t & 31, wave = t >> 5;
    const int wm = wave >> 2, wn = wave & 3;          // wave grid 2 x 4
    const int e = expert_idx[b];
    const float* xb = x + (size_t)b * C_ * HW_;

    union Frag { v16bf v; uint32_t u[8]; };
    v8f zero = {};
    v8f acc[4][2];
    #pragma unroll
    for (int ms = 0; ms < 4; ++ms)
        #pragma unroll
        for (int ns = 0; ns < 2; ++ns) acc[ms][ns] = zero;

    // Per-lane K indices for 16-bit A/B fragment layout (ISA 7.12.2):
    // lanes 0-15 hold K 0..7 & 16..23 (dword pairs), lanes 16-31 hold 8..15 & 24..31.
    const int kh = (lane >> 4) * 8;
    int kidx[8];
    #pragma unroll
    for (int i = 0; i < 8; ++i)
        kidx[i] = (i < 4) ? (kh + 2 * i) : (16 + kh + 2 * (i - 4));

#ifdef HAVE_TDM
    const uint32_t sAh_off = (uint32_t)(uintptr_t)&sAh[0];
    const uint32_t sAl_off = (uint32_t)(uintptr_t)&sAl[0];
#endif

    for (int rs = 0; rs < 9; ++rs) {
        const int r = rs / 3, s = rs % 3;
        const size_t tileoff = (((size_t)e * 9 + rs) * OC_ + mt * 128) * C_;

        for (int c0 = 0; c0 < C_; c0 += 32) {
            __syncthreads();   // previous tile fully consumed

            // ---- stage A (weights): 128 oc x 32 c, bf16 pre-split ----
#ifdef HAVE_TDM
            if (wave == 0) {   // one wave drives both DMA descriptors
                tdm_load_tile_bf16(wq_hi + tileoff + c0, sAh_off);
                tdm_load_tile_bf16(wq_lo + tileoff + c0, sAl_off);
            }
#else
            {
                const uint32_t* gAh = (const uint32_t*)(wq_hi + tileoff);
                const uint32_t* gAl = (const uint32_t*)(wq_lo + tileoff);
                uint32_t* dAh = (uint32_t*)sAh;
                uint32_t* dAl = (uint32_t*)sAl;
                #pragma unroll
                for (int i = 0; i < 8; ++i) {
                    const int idx = i * 256 + t;               // 0..2047 dwords
                    const int row = idx >> 4, pr = idx & 15;
                    const int g = row * 128 + (c0 >> 1) + pr;
                    dAh[idx] = gAh[g];
                    dAl[idx] = gAl[g];
                }
            }
#endif
            // pull next x channel chunk toward cache while we convert this one
            if (c0 + 32 < C_)
                __builtin_prefetch(xb + (size_t)(c0 + 32 + (t & 31)) * HW_ + nt * 128, 0, 1);

            // ---- stage B (im2col of x): 32 c x 128 p, split on the fly ----
            #pragma unroll
            for (int i = 0; i < 16; ++i) {
                const int idx = i * 256 + t;               // 0..4095
                const int cl = idx >> 7, pl = idx & 127;   // p fastest: coalesced
                const int p = nt * 128 + pl;
                float xv = 0.f;
                if (p < HW_) {
                    const int h = p / W_IMG, w = p - h * W_IMG;
                    const int h2 = h + r - 1, w2 = w + s - 1;
                    if ((unsigned)h2 < (unsigned)W_IMG && (unsigned)w2 < (unsigned)W_IMG)
                        xv = xb[(size_t)(c0 + cl) * HW_ + h2 * W_IMG + w2];
                }
                const uint32_t hi = f32_to_bf16(xv);
                const float    lo = xv - bf16hi_to_f32(hi);
                sBh[pl * 32 + cl] = (uint16_t)hi;
                sBl[pl * 32 + cl] = (uint16_t)f32_to_bf16(lo);
            }

#ifdef HAVE_TDM
            if (wave == 0) __builtin_amdgcn_s_wait_tensorcnt(0);
#endif
            __syncthreads();

            // ---- fragments ----
            Frag ah[4], al[4], bh[2], bl[2];
            #pragma unroll
            for (int ms = 0; ms < 4; ++ms) {
                const int row = wm * 64 + ms * 16 + (lane & 15);
                #pragma unroll
                for (int i = 0; i < 8; ++i) {
                    ah[ms].u[i] = *(const uint32_t*)&sAh[row * 32 + kidx[i]];
                    al[ms].u[i] = *(const uint32_t*)&sAl[row * 32 + kidx[i]];
                }
            }
            #pragma unroll
            for (int ns = 0; ns < 2; ++ns) {
                const int col = wn * 32 + ns * 16 + (lane & 15);
                #pragma unroll
                for (int i = 0; i < 8; ++i) {
                    bh[ns].u[i] = *(const uint32_t*)&sBh[col * 32 + kidx[i]];
                    bl[ns].u[i] = *(const uint32_t*)&sBl[col * 32 + kidx[i]];
                }
            }

            // ---- bf16x3 WMMA: 24 v_wmma_f32_16x16x32_bf16 per k-step ----
            #pragma unroll
            for (int ms = 0; ms < 4; ++ms)
                #pragma unroll
                for (int ns = 0; ns < 2; ++ns) {
                    acc[ms][ns] = __builtin_amdgcn_wmma_f32_16x16x32_bf16(
                        false, ah[ms].v, false, bh[ns].v, (short)0, acc[ms][ns], false, false);
                    acc[ms][ns] = __builtin_amdgcn_wmma_f32_16x16x32_bf16(
                        false, ah[ms].v, false, bl[ns].v, (short)0, acc[ms][ns], false, false);
                    acc[ms][ns] = __builtin_amdgcn_wmma_f32_16x16x32_bf16(
                        false, al[ms].v, false, bh[ns].v, (short)0, acc[ms][ns], false, false);
                }
        }
    }

    // ---- epilogue: + expert bias, masked store (C/D layout 7.12.2) ----
    const int mhi = (lane >> 4) * 8;
    #pragma unroll
    for (int ms = 0; ms < 4; ++ms) {
        #pragma unroll
        for (int v = 0; v < 8; ++v) {
            const int oc = mt * 128 + wm * 64 + ms * 16 + v + mhi;
            const float bias = bias_w[e * OC_ + oc];
            float* orow = out + ((size_t)b * OC_ + oc) * HW_;
            #pragma unroll
            for (int ns = 0; ns < 2; ++ns) {
                const int p = nt * 128 + wn * 32 + ns * 16 + (lane & 15);
                if (p < HW_) orow[p] = acc[ms][ns][v] + bias;
            }
        }
    }
}

// ---------------------------------------------------------------------------
extern "C" void kernel_launch(void* const* d_in, const int* in_sizes, int n_in,
                              void* d_out, int out_size, void* d_ws, size_t ws_size,
                              hipStream_t stream) {
    const float* x      = (const float*)d_in[0];
    const float* w_fc1  = (const float*)d_in[1];
    const float* w_fc2  = (const float*)d_in[2];
    const float* b_fc2  = (const float*)d_in[3];
    const float* alpha  = (const float*)d_in[4];
    const float* weight = (const float*)d_in[5];
    const float* bias_w = (const float*)d_in[6];

    float* out = (float*)d_out;
    float* raw = out + RAW_OFF;   // tuple output: (out, raw) concatenated

    // workspace: [0,256) expert ids; then bf16 hi / lo split weights (9.44 MB)
    char* ws = (char*)d_ws;
    int* expert = (int*)ws;
    uint16_t* wq_hi = (uint16_t*)(ws + 256);
    uint16_t* wq_lo = wq_hi + (size_t)KEXP * MSIZE;

    attn_kernel<<<B_, 256, 0, stream>>>(x, w_fc1, w_fc2, b_fc2, raw, expert);

    const int total = KEXP * MSIZE;
    quant_split_kernel<<<(total + 255) / 256, 256, 0, stream>>>(weight, alpha, wq_hi, wq_lo);

    conv_wmma_kernel<<<dim3(25, 2, B_), 256, 0, stream>>>(x, wq_hi, wq_lo,
                                                          bias_w, expert, out);
}
